// QuantizedLinear_22849226015470
// MI455X (gfx1250) — matrix operand order repaired and checked by reading, not compiled, not run on hardware
//
#include <hip/hip_runtime.h>
#include <hip/hip_bf16.h>

// ---------------------------------------------------------------------------
// QuantizedLinear (QTIP trellis) for MI455X / gfx1250
//   out[8192,4096] = fwht128(x) @ W^T,  W decoded from 16-bit trellis + LUT
// Plan: (1) FWHT rotate x -> bf16 ws   (2) decode W -> bf16 ws (fits in L2)
//       (3) bf16 WMMA GEMM: 256x128 block tile, 64x64 per wave (16 WMMA/iter),
//         double-buffered GLOBAL_LOAD_ASYNC_TO_LDS_B128 staging (ASYNCcnt),
//         chunk displacement via the instruction IOFFSET (applies to both the
//         LDS dest and global src per CDNA5 ISA 8.2 -- no per-chunk VALU adds).
// ---------------------------------------------------------------------------

typedef __attribute__((ext_vector_type(16))) __bf16 bf16x16;
typedef __attribute__((ext_vector_type(8)))  float  f32x8;

#define IN_F  4096
#define OUT_F 4096
#define M_TOT 8192          // 4 * 2048 rows
#define BK    32            // K per WMMA step
#define NITER (IN_F / BK)   // 128
#define BUFSZ 24576         // A tile 16 KiB + B tile 8 KiB

__device__ __forceinline__ unsigned short f2bf(float f) {
    unsigned u = __float_as_uint(f);
    u += 0x7FFFu + ((u >> 16) & 1u);     // round-to-nearest-even
    return (unsigned short)(u >> 16);
}

// ---------------------------------------------------------------------------
// Kernel 1: per-128 group Walsh-Hadamard rotate + bf16 convert.
// ---------------------------------------------------------------------------
__global__ __launch_bounds__(128)
void hadamard_rotate_kernel(const float* __restrict__ x,
                            unsigned short* __restrict__ xr) {
    __shared__ float buf[128];
    const int tid = threadIdx.x;
    const long long g = blockIdx.x;           // group id: 0 .. 8192*32-1
    float v = x[g * 128 + tid];
    #pragma unroll
    for (int s = 0; s < 7; ++s) {
        buf[tid] = v;
        __syncthreads();
        const int bit = 1 << s;
        const float o = buf[tid ^ bit];
        v = (tid & bit) ? (o - v) : (v + o);
        __syncthreads();
    }
    xr[g * 128 + tid] = f2bf(v * 0.08838834764831845f);   // 1/sqrt(128)
}

// ---------------------------------------------------------------------------
// Kernel 2: trellis decode -> bf16 weight matrix W[n][k] (row-major).
// state(i) = 16-bit window at bit 2*i of the circular 512-bit stream.
// ---------------------------------------------------------------------------
__global__ __launch_bounds__(256)
void decode_weights_kernel(const int*   __restrict__ trellis,
                           const float* __restrict__ tlut,
                           const float* __restrict__ scales,
                           unsigned short* __restrict__ wb) {
    const int t = blockIdx.x;           // tile 0..65535
    const int i = threadIdx.x;          // element 0..255 within tile
    const int p  = i << 1;              // bit position (K=2)
    const int kw = p >> 4;              // word index
    const int sh = p & 15;              // shift within word
    const unsigned w0 = (unsigned)trellis[t * 32 + kw] & 0xFFFFu;
    const unsigned w1 = (unsigned)trellis[t * 32 + ((kw + 1) & 31)] & 0xFFFFu;
    const unsigned state = ((w0 << sh) | (w1 >> (16 - sh))) & 0xFFFFu;
    const float val = tlut[state];                                       // V == 1
    const int r = i >> 4, c = i & 15;
    const int row = ((t >> 8) << 4) + r;        // out_feature
    const int col = ((t & 255) << 4) + c;       // in_feature
    const long long flat = (long long)row * IN_F + col;
    wb[flat] = f2bf(val * scales[flat >> 7]);   // groups of 128 along in_features
}

// ---------------------------------------------------------------------------
// Async global -> LDS with immediate chunk offset (ASYNCcnt-tracked).
// IOFFSET is added to BOTH lds dest and global src (ISA 08 §4.4) which is
// exactly the contiguous-chunk staging pattern used here.
// ---------------------------------------------------------------------------
template <int OFF>
__device__ __forceinline__ void async_b128(unsigned ldsOff, const void* g) {
    asm volatile("global_load_async_to_lds_b128 %0, %1, off offset:%2"
                 :: "v"(ldsOff), "v"(g), "n"(OFF) : "memory");
}

// ---------------------------------------------------------------------------
// Kernel 3: bf16 WMMA GEMM.  C[m,n] = sum_k A[m,k] * B[n,k]
//   block: 256 threads (8 wave32), 256x128 output tile, BK = 32
//   wave (wm,wn): 64x64 sub-tile = 4x4 WMMA accumulators
//   LDS double buffer: [A 256x32 | B 128x32] x 2 = 48 KiB
//   Per wave per tile: 6 async-to-LDS loads -> s_wait_asynccnt 6 retires
//   exactly the older of the two in-flight tiles (async loads finish in order).
// ---------------------------------------------------------------------------
union FragBF { uint4 q[2]; bf16x16 v; };

__device__ __forceinline__ void stage_tile(unsigned aOff, unsigned bOff,
                                           const char* gA, const char* gB) {
    async_b128< 0>(aOff, gA);
    async_b128<16>(aOff, gA);
    async_b128<32>(aOff, gA);
    async_b128<48>(aOff, gA);
    async_b128< 0>(bOff, gB);
    async_b128<16>(bOff, gB);
}

__global__ __launch_bounds__(256)
void gemm_bf16_wmma_kernel(const unsigned short* __restrict__ A,   // [M_TOT][IN_F] bf16
                           const unsigned short* __restrict__ B,   // [OUT_F][IN_F] bf16
                           float* __restrict__ C) {                // [M_TOT][OUT_F]
    __shared__ __align__(16) unsigned char smem[2 * BUFSZ];

    const int tid  = threadIdx.x;
    const int lane = tid & 31;
    const int w    = tid >> 5;
    const int wmRow = (w >> 1) * 64;    // 0,64,128,192
    const int wnCol = (w & 1) * 64;     // 0,64
    const int m0   = blockIdx.y * 256;
    const int n0   = blockIdx.x * 128;

    const int mlo   = lane & 15;
    const int khalf = lane >> 4;

    // LDS byte offsets (low 32 bits of generic shared address = LDS offset)
    const unsigned sbase = (unsigned)(unsigned long long)(const void*)smem;
    const unsigned aOff  = sbase + tid * 64;                       // A: row tid, 64B/row
    const unsigned bOff  = sbase + 16384 + (tid >> 1) * 64 + (tid & 1) * 32;

    // per-thread global staging pointers (advance 64 B per K-tile)
    const char* gA = (const char*)(A + (size_t)(m0 + tid) * IN_F);
    const char* gB = (const char*)(B + (size_t)(n0 + (tid >> 1)) * IN_F) + (tid & 1) * 32;

    f32x8 acc[4][4] = {};

    // ---- prologue: prefetch tiles 0 and 1 (6 async ops each per wave) ----
    stage_tile(aOff,         bOff,         gA,      gB);
    stage_tile(aOff + BUFSZ, bOff + BUFSZ, gA + 64, gB + 64);
    gA += 128; gB += 128;

    for (int i = 0; i < NITER; ++i) {
        // retire the older in-flight tile (in-order async completion)
        if (i + 1 < NITER) asm volatile("s_wait_asynccnt 6" ::: "memory");
        else               asm volatile("s_wait_asynccnt 0" ::: "memory");
        __syncthreads();

        const unsigned char* lA = smem + (i & 1) * BUFSZ;
        const unsigned char* lB = lA + 16384;

        // A fragments: lane = khalf*16+m -> row m, 16B chunks at khalf*16, 32+khalf*16
        FragBF fa[4];
        #pragma unroll
        for (int mi = 0; mi < 4; ++mi) {
            const int row = wmRow + mi * 16 + mlo;
            fa[mi].q[0] = *(const uint4*)(lA + row * 64 +      khalf * 16);
            fa[mi].q[1] = *(const uint4*)(lA + row * 64 + 32 + khalf * 16);
        }
        // B fragments: lane = khalf*16+n -> W[n][k], 16B chunks at khalf*16, 32+khalf*16
        #pragma unroll
        for (int ni = 0; ni < 4; ++ni) {
            FragBF fb;
            const int row = wnCol + ni * 16 + mlo;
            const uint4* p = (const uint4*)(lB + row * 64 + khalf * 16);
            fb.q[0] = p[0];
            fb.q[1] = p[2];
            #pragma unroll
            for (int mi = 0; mi < 4; ++mi)
                acc[mi][ni] = __builtin_amdgcn_wmma_f32_16x16x32_bf16(
                    false, fa[mi].v, false, fb.v,
                    (short)0, acc[mi][ni], false, false);
        }
        __syncthreads();   // everyone done reading buf (i&1) before overwrite

        if (i + 2 < NITER) {
            stage_tile(aOff + (unsigned)(i & 1) * BUFSZ,
                       bOff + (unsigned)(i & 1) * BUFSZ, gA, gB);
            gA += 64; gB += 64;
        }
    }

    // ---- epilogue: C/D layout = VGPR v, lane L -> M = (L>>4)*8 + v, N = L&15
    #pragma unroll
    for (int mi = 0; mi < 4; ++mi) {
        const int rbase = m0 + wmRow + mi * 16 + khalf * 8;
        #pragma unroll
        for (int ni = 0; ni < 4; ++ni) {
            const int col = n0 + wnCol + ni * 16 + mlo;
            #pragma unroll
            for (int v = 0; v < 8; ++v)
                C[(size_t)(rbase + v) * OUT_F + col] = acc[mi][ni][v];
        }
    }
}

// ---------------------------------------------------------------------------
extern "C" void kernel_launch(void* const* d_in, const int* in_sizes, int n_in,
                              void* d_out, int out_size, void* d_ws, size_t ws_size,
                              hipStream_t stream) {
    const float* input   = (const float*)d_in[0];   // [4,2048,4096] fp32
    const int*   trellis = (const int*)  d_in[1];   // [65536,32] int32
    const float* tlut    = (const float*)d_in[2];   // [65536,1] fp32
    const float* scales  = (const float*)d_in[3];   // [131072,1] fp32
    float* out = (float*)d_out;                     // [4,2048,4096] fp32

    // workspace layout: x_rot bf16 (64 MiB region) | W bf16 (32 MiB)
    unsigned short* xrot = (unsigned short*)d_ws;
    unsigned short* wbf  = xrot + (size_t)M_TOT * IN_F;

    // 1) Hadamard rotate -> bf16
    hadamard_rotate_kernel<<<(M_TOT * (IN_F / 128)), 128, 0, stream>>>(input, xrot);

    // 2) Trellis decode -> bf16 weights (33 MB, L2-resident)
    decode_weights_kernel<<<(OUT_F / 16) * (IN_F / 16), 256, 0, stream>>>(
        trellis, tlut, scales, wbf);

    // 3) Tensor-core GEMM: 256x128 block tiles
    dim3 grid(OUT_F / 128, M_TOT / 256);   // (32, 32)
    gemm_bf16_wmma_kernel<<<grid, 256, 0, stream>>>(xrot, wbf, out);
}
